// SlimMambaBlock_84567906058585
// MI455X (gfx1250) — matrix operand order repaired and checked by reading, not compiled, not run on hardware
//
#include <hip/hip_runtime.h>
#include <hip/hip_bf16.h>

// ---------------------------------------------------------------------------
// SlimMambaBlock on gfx1250 (MI455X): bf16 WMMA GEMMs (async-LDS double
// buffered, 64x64 wave tiles) + elementwise scan.
// B=8, K=2048, D=1024, INNER=2048. M = B*K = 16384 tokens.
// ---------------------------------------------------------------------------

typedef __attribute__((ext_vector_type(16))) __bf16 v16bf;
typedef __attribute__((ext_vector_type(8)))  float  v8f;

#define M_TOK   16384
#define D_DIM   1024
#define INNER_D 2048

#if defined(__has_builtin)
#  if __has_builtin(__builtin_amdgcn_global_load_async_to_lds_b128)
#    define HAVE_ASYNC_LDS 1
#  endif
#endif
#ifndef HAVE_ASYNC_LDS
#  define HAVE_ASYNC_LDS 0
#endif

#if defined(__has_builtin) && __has_builtin(__builtin_amdgcn_s_wait_asynccnt)
#  define WAIT_ASYNC(n) __builtin_amdgcn_s_wait_asynccnt(n)
#else
#  define WAIT_ASYNC(n) asm volatile("s_wait_asynccnt %0" :: "i"(n) : "memory")
#endif

// native clang vector (NOT HIP_vector_type) — matches builtin parameter type
typedef int nint4 __attribute__((vector_size(16)));
typedef __attribute__((address_space(1))) nint4 as1_nint4;   // global
typedef __attribute__((address_space(3))) nint4 as3_nint4;   // LDS

union FragAB {
    v16bf v;
    int4  q[2];
};

__device__ __forceinline__ float fsilu(float x)   { return x / (1.0f + __expf(-x)); }
__device__ __forceinline__ float fsigmoid(float x){ return 1.0f / (1.0f + __expf(-x)); }

// 16-byte global -> LDS copy: async DMA path on CDNA5, reg-staged fallback.
__device__ __forceinline__ void copy16_g2l(const __bf16* g, __bf16* l)
{
#if HAVE_ASYNC_LDS
    __builtin_amdgcn_global_load_async_to_lds_b128(
        (as1_nint4*)g, (as3_nint4*)l, 0, 0);
#else
    *(int4*)l = *(const int4*)g;
#endif
}

// ---------------------------------------------------------------------------
// fp32 -> bf16 weight conversion (n divisible by 4)
// ---------------------------------------------------------------------------
__global__ void __launch_bounds__(256)
cvt_f32_bf16_kernel(const float* __restrict__ in, __bf16* __restrict__ out, int n)
{
    int i = (blockIdx.x * 256 + threadIdx.x) * 4;
    if (i < n) {
        float4 v = *(const float4*)(in + i);
        out[i + 0] = (__bf16)v.x;
        out[i + 1] = (__bf16)v.y;
        out[i + 2] = (__bf16)v.z;
        out[i + 3] = (__bf16)v.w;
    }
}

// ---------------------------------------------------------------------------
// RMSNorm: one block (256 thr) per token row of D=1024. h = x*rsqrt(mean+eps)*w
// ---------------------------------------------------------------------------
__global__ void __launch_bounds__(256)
rmsnorm_kernel(const float* __restrict__ x, const float* __restrict__ w,
               __bf16* __restrict__ h)
{
    __shared__ float red[8];
    const int row = blockIdx.x;
    const int tid = threadIdx.x;

    float4 v = ((const float4*)(x + (size_t)row * D_DIM))[tid];
    float ss = v.x * v.x + v.y * v.y + v.z * v.z + v.w * v.w;

    #pragma unroll
    for (int o = 16; o > 0; o >>= 1) ss += __shfl_xor(ss, o, 32);
    if ((tid & 31) == 0) red[tid >> 5] = ss;
    __syncthreads();
    if (tid < 8) {
        float t = red[tid];
        t += __shfl_xor(t, 1, 32);
        t += __shfl_xor(t, 2, 32);
        t += __shfl_xor(t, 4, 32);
        if (tid == 0) red[0] = t;
    }
    __syncthreads();
    const float scale = rsqrtf(red[0] * (1.0f / (float)D_DIM) + 1e-5f);

    float4 wv = ((const float4*)w)[tid];
    __bf16* hp = h + (size_t)row * D_DIM + tid * 4;
    hp[0] = (__bf16)(v.x * scale * wv.x);
    hp[1] = (__bf16)(v.y * scale * wv.y);
    hp[2] = (__bf16)(v.z * scale * wv.z);
    hp[3] = (__bf16)(v.w * scale * wv.w);
}

// ---------------------------------------------------------------------------
// bf16 WMMA GEMM:  C[M,N] = A[M,Kd] * Bw[N,Kd]^T  (+ epilogue)
//   EPI 1: silu-split (GEMM1):  col<INNER -> u bf16 ; col>=INNER -> silu(z) bf16
//   EPI 2: bias+sigmoid (GEMM2) -> lam bf16
//   EPI 3: residual add (GEMM3) -> outf = xres + acc (fp32)
// Block tile 128x256x32, 8 waves (2 row-groups x 4 col-groups), wave = 64x64
// (16 WMMAs / 16 fragment loads per k-step). Double-buffered async LDS.
// ---------------------------------------------------------------------------
template <int EPI>
__global__ void __launch_bounds__(256)
gemm_bf16_wmma(const __bf16* __restrict__ A, const __bf16* __restrict__ Bw,
               const float* __restrict__ bias, const float* __restrict__ xres,
               float* __restrict__ outf,
               __bf16* __restrict__ out0, __bf16* __restrict__ out1,
               int N, int Kd)
{
    constexpr int BM = 128, BN = 256, BK = 32;
    __shared__ __bf16 sA[2][BM * BK];   // 2 x 8 KB
    __shared__ __bf16 sB[2][BN * BK];   // 2 x 16 KB

    const int tid  = threadIdx.x;
    const int lane = tid & 31;
    const int wave = tid >> 5;
    const int wr   = wave >> 2;          // 0..1 : 64-row group
    const int wc   = wave & 3;           // 0..3 : 64-col group
    const int half = lane >> 4;          // hi/lo half of wave
    const int l16  = lane & 15;
    const int bm   = blockIdx.y * BM;
    const int bn   = blockIdx.x * BN;

    // per-thread staging chunk: 16B at (row = tid>>2 [+64i], col = (tid&3)*8)
    const int cRow = tid >> 2;
    const int cCol = (tid & 3) * 8;
    const __bf16* Ag = A  + (size_t)(bm + cRow) * Kd + cCol;
    const __bf16* Bg = Bw + (size_t)(bn + cRow) * Kd + cCol;
    const int ldsOff = cRow * BK + cCol;

    const v8f vzero = {0.f, 0.f, 0.f, 0.f, 0.f, 0.f, 0.f, 0.f};
    v8f acc[4][4];
    #pragma unroll
    for (int i = 0; i < 4; ++i)
        #pragma unroll
        for (int j = 0; j < 4; ++j) acc[i][j] = vzero;

    const int nk = Kd / BK;

    // stage tile kt into buffer buf (6 x 16B per thread: 2 for A, 4 for B)
    auto stage = [&](int buf, int k0) {
        #pragma unroll
        for (int i = 0; i < 2; ++i)
            copy16_g2l(Ag + (size_t)i * 64 * Kd + k0, &sA[buf][ldsOff + i * 64 * BK]);
        #pragma unroll
        for (int i = 0; i < 4; ++i)
            copy16_g2l(Bg + (size_t)i * 64 * Kd + k0, &sB[buf][ldsOff + i * 64 * BK]);
    };

    stage(0, 0);

    for (int kt = 0; kt < nk; ++kt) {
        const int cur = kt & 1;
        if (kt + 1 < nk) {
            stage(1 - cur, (kt + 1) * BK);  // WAR-safe: barrier at end of kt-1
            WAIT_ASYNC(6);                  // this wave's cur-tile copies retired
        } else {
            WAIT_ASYNC(0);
        }
        __syncthreads();                    // all waves: cur tile visible in LDS

        // ---- fragments per ISA 7.12.2 lane layout ----
        FragAB af[4], bfr[4];
        const char* baseA = (const char*)sA[cur];
        const char* baseB = (const char*)sB[cur];
        #pragma unroll
        for (int t = 0; t < 4; ++t) {
            const char* ra = baseA + (wr * 64 + t * 16 + l16) * 64;
            af[t].q[0] = *(const int4*)(ra + half * 16);        // K 0..7 / 8..15
            af[t].q[1] = *(const int4*)(ra + 32 + half * 16);   // K 16..23 / 24..31
        }
        #pragma unroll
        for (int t = 0; t < 4; ++t) {
            const char* rb = baseB + (wc * 64 + t * 16 + l16) * 64;
            bfr[t].q[0] = *(const int4*)(rb + half * 16);       // K 0..7 / 8..15
            bfr[t].q[1] = *(const int4*)(rb + 32 + half * 16);  // K 16..23 / 24..31
        }

        #pragma unroll
        for (int i = 0; i < 4; ++i)
            #pragma unroll
            for (int j = 0; j < 4; ++j)
                acc[i][j] = __builtin_amdgcn_wmma_f32_16x16x32_bf16(
                    false, af[i].v, false, bfr[j].v,
                    (short)0, acc[i][j], false, false);

        __syncthreads();                    // all waves done reading cur
    }

    // ---- epilogue: lane l16 = column, VGPR r + 8*half = row within tile ----
    #pragma unroll
    for (int i = 0; i < 4; ++i) {
        #pragma unroll
        for (int j = 0; j < 4; ++j) {
            const int col = bn + wc * 64 + j * 16 + l16;
            #pragma unroll
            for (int r = 0; r < 8; ++r) {
                const int row = bm + wr * 64 + i * 16 + half * 8 + r;
                float v = acc[i][j][r];
                if (EPI == 1) {
                    float sv = fsilu(v);
                    if (col < INNER_D)
                        out0[(size_t)row * INNER_D + col] = (__bf16)sv;
                    else
                        out1[(size_t)row * INNER_D + (col - INNER_D)] = (__bf16)sv;
                } else if (EPI == 2) {
                    float sg = fsigmoid(v + bias[col]);
                    out0[(size_t)row * INNER_D + col] = (__bf16)sg;
                } else { // EPI == 3
                    outf[(size_t)row * N + col] = xres[(size_t)row * N + col] + v;
                }
            }
        }
    }
}

// ---------------------------------------------------------------------------
// Sequential selective scan over K per (batch, channel). s_eff may alias lam
// (same index read-before-write per thread; threads own disjoint channels).
// ---------------------------------------------------------------------------
__global__ void __launch_bounds__(256)
scan_kernel(const __bf16* lam, const __bf16* u, const __bf16* sz,
            __bf16* s_eff, int Ksteps)
{
    const int idx = blockIdx.x * 256 + threadIdx.x;   // b*INNER + j
    const int b = idx >> 11;
    const int j = idx & (INNER_D - 1);
    size_t p = ((size_t)b * Ksteps) * INNER_D + j;

    float s = 0.0f;
    for (int k = 0; k < Ksteps; ++k, p += INNER_D) {
        const float l  = (float)lam[p];
        const float uu = (float)u[p];
        const float zz = (float)sz[p];
        s = l * s + (1.0f - l) * uu;
        s_eff[p] = (__bf16)(s * zz);
    }
}

// ---------------------------------------------------------------------------
// Launch
// ---------------------------------------------------------------------------
extern "C" void kernel_launch(void* const* d_in, const int* in_sizes, int n_in,
                              void* d_out, int out_size, void* d_ws, size_t ws_size,
                              hipStream_t stream)
{
    const float* x      = (const float*)d_in[0];   // (8,2048,1024)
    const float* w_norm = (const float*)d_in[1];   // (1024)
    const float* W_in   = (const float*)d_in[2];   // (4096,1024)
    const float* W_dt   = (const float*)d_in[3];   // (2048,2048)
    const float* b_dt   = (const float*)d_in[4];   // (2048)
    const float* W_out  = (const float*)d_in[5];   // (1024,2048)
    float* out = (float*)d_out;

    char* ws = (char*)d_ws;
    __bf16* Win_bf  = (__bf16*)(ws);                        //  8 MB
    __bf16* Wdt_bf  = (__bf16*)(ws + (8ull  << 20));        //  8 MB
    __bf16* Wout_bf = (__bf16*)(ws + (16ull << 20));        //  4 MB
    __bf16* hbuf    = (__bf16*)(ws + (20ull << 20));        // 32 MB
    __bf16* ubuf    = (__bf16*)(ws + (52ull << 20));        // 64 MB
    __bf16* szbuf   = (__bf16*)(ws + (116ull << 20));       // 64 MB
    __bf16* lambuf  = (__bf16*)(ws + (180ull << 20));       // 64 MB (total 244 MB)
    __bf16* sbuf    = lambuf;                               // aliased: read-before-write

    // weight conversions
    cvt_f32_bf16_kernel<<<(4096 * 1024) / 1024, 256, 0, stream>>>(W_in,  Win_bf,  4096 * 1024);
    cvt_f32_bf16_kernel<<<(2048 * 2048) / 1024, 256, 0, stream>>>(W_dt,  Wdt_bf,  2048 * 2048);
    cvt_f32_bf16_kernel<<<(1024 * 2048) / 1024, 256, 0, stream>>>(W_out, Wout_bf, 1024 * 2048);

    // RMSNorm
    rmsnorm_kernel<<<M_TOK, 256, 0, stream>>>(x, w_norm, hbuf);

    // GEMM1: uv = h @ W_in^T  (M=16384, N=4096, K=1024), silu-split epilogue
    gemm_bf16_wmma<1><<<dim3(4096 / 256, M_TOK / 128), 256, 0, stream>>>(
        hbuf, Win_bf, nullptr, nullptr, nullptr, ubuf, szbuf, 4096, 1024);

    // GEMM2: lam = sigmoid(u @ W_dt^T + b)  (M=16384, N=2048, K=2048)
    gemm_bf16_wmma<2><<<dim3(2048 / 256, M_TOK / 128), 256, 0, stream>>>(
        ubuf, Wdt_bf, b_dt, nullptr, nullptr, lambuf, nullptr, 2048, 2048);

    // sequential scan over K, fused with * silu(z)
    scan_kernel<<<(8 * INNER_D) / 256, 256, 0, stream>>>(lambuf, ubuf, szbuf, sbuf, 2048);

    // GEMM3: out = x + s_eff @ W_out^T  (M=16384, N=1024, K=2048)
    gemm_bf16_wmma<3><<<dim3(1024 / 256, M_TOK / 128), 256, 0, stream>>>(
        sbuf, Wout_bf, nullptr, x, out, nullptr, nullptr, 1024, 2048);
}
